// Attention_22497038696616
// MI455X (gfx1250) — matrix operand order repaired and checked by reading, not compiled, not run on hardware
//
#include <hip/hip_runtime.h>
#include <hip/hip_bf16.h>

typedef _Float16 v8h  __attribute__((ext_vector_type(8)));
typedef _Float16 v16h __attribute__((ext_vector_type(16)));
typedef float    v8f  __attribute__((ext_vector_type(8)));

#define B_   8
#define N_   1024
#define H_   12
#define HD_  64
#define C_   768
#define SCALE_ 0.125f            // 64^-0.5
#define EPS_   1e-6f

__device__ __forceinline__ v16h cat8(v8h a, v8h b) {
  return __builtin_shufflevector(a, b, 0,1,2,3,4,5,6,7,8,9,10,11,12,13,14,15);
}

// A operand (16xK=32, f16): lane g (0..15) holds row r0+g; hi=lane>>4.
// elements 0..7  = K k0 + hi*8 + {0..7}
// elements 8..15 = K k0 + 16 + hi*8 + {0..7}
__device__ __forceinline__ v16h load_afrag(const _Float16* p, int ld, int r0, int k0,
                                           int g, int hi) {
  const _Float16* r = p + (size_t)(r0 + g) * ld + k0 + hi * 8;
  v8h c0 = *(const v8h*)r;
  v8h c1 = *(const v8h*)(r + 16);
  return cat8(c0, c1);
}

// B operand (K=32 x 16) where B[k][n] = W[n0+n][k0+k], W row-major (K contiguous).
// lane g -> n; elements 0..7 = K k0 + hi*8 + {0..7}, elements 8..15 = K k0 + 16 + hi*8 + {0..7}
__device__ __forceinline__ v16h load_bfrag(const _Float16* p, int ld, int n0, int k0,
                                           int g, int hi) {
  const _Float16* r = p + (size_t)(n0 + g) * ld + k0 + hi * 8;
  v8h c0 = *(const v8h*)r;
  v8h c1 = *(const v8h*)(r + 16);
  return cat8(c0, c1);
}

#define WMMA_F16(a, b, c) \
  __builtin_amdgcn_wmma_f32_16x16x32_f16(false, (a), false, (b), (short)0, (c), false, false)

// ---------------------------------------------------------------- convert f32 -> f16
__global__ void cvt_f32_f16(const float* __restrict__ in, _Float16* __restrict__ out, int n) {
  for (int i = blockIdx.x * blockDim.x + threadIdx.x; i < n; i += gridDim.x * blockDim.x)
    out[i] = (_Float16)in[i];
}

// ---------------------------------------------------------------- QKV GEMM
// C[8192,2304] = xh[8192,768] @ wqkv[2304,768]^T ; scatter into q,k (B,H,N,64) and vT (B,H,64,N)
__global__ __launch_bounds__(256) void qkv_gemm(const _Float16* __restrict__ A,
                                                const _Float16* __restrict__ W,
                                                _Float16* __restrict__ qh,
                                                _Float16* __restrict__ kh,
                                                _Float16* __restrict__ vT) {
  const int tid = threadIdx.x, wave = tid >> 5, lane = tid & 31;
  const int g = lane & 15, hi = lane >> 4;
  const int mw = wave >> 1, nw = wave & 1;
  const int row0 = blockIdx.x * 128 + mw * 32;
  const int col0 = blockIdx.y * 128 + nw * 64;

  v8f acc[2][4] = {};
  for (int k0 = 0; k0 < C_; k0 += 32) {
    if (k0 + 32 < C_)
      __builtin_prefetch(A + (size_t)(row0 + g) * C_ + k0 + 32, 0, 0);
    v16h a0 = load_afrag(A, C_, row0,      k0, g, hi);
    v16h a1 = load_afrag(A, C_, row0 + 16, k0, g, hi);
#pragma unroll
    for (int t = 0; t < 4; ++t) {
      v16h b = load_bfrag(W, C_, col0 + t * 16, k0, g, hi);
      acc[0][t] = WMMA_F16(a0, b, acc[0][t]);
      acc[1][t] = WMMA_F16(a1, b, acc[1][t]);
    }
  }
#pragma unroll
  for (int ms = 0; ms < 2; ++ms)
#pragma unroll
    for (int t = 0; t < 4; ++t) {
      const int col = col0 + t * 16 + g;          // 0..2303
      const int s   = col / C_;                    // 0=q 1=k 2=v
      const int rem = col % C_;
      const int h   = rem >> 6, d = rem & 63;
#pragma unroll
      for (int j = 0; j < 8; ++j) {
        const int row = row0 + ms * 16 + j + 8 * hi;   // b*1024+n
        const int b = row >> 10, n = row & 1023;
        const int bh = b * H_ + h;
        const _Float16 val = (_Float16)acc[ms][t][j];
        if (s == 0)      qh[((size_t)bh * N_ + n) * HD_ + d] = val;
        else if (s == 1) kh[((size_t)bh * N_ + n) * HD_ + d] = val;
        else             vT[((size_t)bh * HD_ + d) * N_ + n] = val;
      }
    }
}

// ---------------------------------------------------------------- V column sums (per b,h,d)
__global__ void vsum_kernel(const _Float16* __restrict__ vT, float* __restrict__ vsum) {
  const int bh = blockIdx.x, d = threadIdx.x;      // 96 blocks x 64 threads
  const v8h* p = (const v8h*)(vT + ((size_t)bh * HD_ + d) * N_);
  float s = 0.f;
  for (int i = 0; i < N_ / 8; ++i) {
    v8h v = p[i];
#pragma unroll
    for (int e = 0; e < 8; ++e) s += (float)v[e];
  }
  vsum[bh * HD_ + d] = s;
}

// ---------------------------------------------------------------- fused flash attention
// one wave per 16-query block of one (b,h); streams 32 keys/iter.
__global__ __launch_bounds__(256) void attn_kernel(const _Float16* __restrict__ qh,
                                                   const _Float16* __restrict__ kh,
                                                   const _Float16* __restrict__ vT,
                                                   const float* __restrict__ policy,
                                                   const float* __restrict__ vsum,
                                                   _Float16* __restrict__ oh) {
  __shared__ __align__(32) _Float16 lds[8][16 * 32];
  const int tid = threadIdx.x, wave = tid >> 5, lane = tid & 31;
  const int g = lane & 15, hi = lane >> 4;
  const int idx = blockIdx.x * 8 + wave;           // 6144 row-block tasks
  const int bh = idx >> 6;
  const int row0 = (idx & 63) * 16;
  const int b = bh / H_, h = bh % H_;

  const _Float16* Q = qh + (size_t)bh * N_ * HD_;
  const _Float16* K = kh + (size_t)bh * N_ * HD_;
  const _Float16* V = vT + (size_t)bh * HD_ * N_;
  const float* pol = policy + (size_t)b * N_;
  _Float16* P = lds[wave];

  const v16h aq0 = load_afrag(Q, HD_, row0, 0, g, hi);
  const v16h aq1 = load_afrag(Q, HD_, row0, 32, g, hi);

  v8f oacc[4] = {};
  float m[8], l[8];
#pragma unroll
  for (int j = 0; j < 8; ++j) { m[j] = -1e30f; l[j] = 0.f; }

  for (int kb = 0; kb < N_; kb += 32) {
    // S = (Q * scale) @ K^T for 32 keys
    v16h bk00 = load_bfrag(K, HD_, kb,      0,  g, hi);
    v16h bk01 = load_bfrag(K, HD_, kb,      32, g, hi);
    v16h bk10 = load_bfrag(K, HD_, kb + 16, 0,  g, hi);
    v16h bk11 = load_bfrag(K, HD_, kb + 16, 32, g, hi);
    v8f s0 = {}, s1 = {};
    s0 = WMMA_F16(aq0, bk00, s0);  s0 = WMMA_F16(aq1, bk01, s0);
    s1 = WMMA_F16(aq0, bk10, s1);  s1 = WMMA_F16(aq1, bk11, s1);

    const float p0c = pol[kb + g];
    const float p1c = pol[kb + 16 + g];

#pragma unroll
    for (int j = 0; j < 8; ++j) {
      const int r = row0 + j + 8 * hi;
      float x0 = s0[j] * SCALE_;
      float x1 = s1[j] * SCALE_;
      float tm = fmaxf(x0, x1);
#pragma unroll
      for (int off = 1; off < 16; off <<= 1) tm = fmaxf(tm, __shfl_xor(tm, off, 16));
      const float nm = fmaxf(m[j], tm);
      const float f = __expf(m[j] - nm);
      m[j] = nm;
      const float w0 = (kb + g == r) ? 1.f : p0c;           // policy + (1-p)*eye
      const float w1 = (kb + 16 + g == r) ? 1.f : p1c;
      const float e0 = __expf(x0 - nm) * w0;
      const float e1 = __expf(x1 - nm) * w1;
      float rs = e0 + e1;
#pragma unroll
      for (int off = 1; off < 16; off <<= 1) rs += __shfl_xor(rs, off, 16);
      l[j] = l[j] * f + rs;
#pragma unroll
      for (int t = 0; t < 4; ++t) oacc[t][j] *= f;
      P[(j + 8 * hi) * 32 + g]      = (_Float16)e0;          // C-layout store
      P[(j + 8 * hi) * 32 + 16 + g] = (_Float16)e1;
    }

    // P back out of LDS in A-operand layout (16x32 over keys)
    v8h c0 = *(const v8h*)&P[g * 32 + hi * 8];
    v8h c1 = *(const v8h*)&P[g * 32 + 16 + hi * 8];
    v16h ap = cat8(c0, c1);
#pragma unroll
    for (int t = 0; t < 4; ++t) {
      v16h bv = load_bfrag(V, N_, t * 16, kb, g, hi);        // contiguous via vT
      oacc[t] = WMMA_F16(ap, bv, oacc[t]);
    }
  }

  const float epsn = EPS_ / (float)N_;
#pragma unroll
  for (int t = 0; t < 4; ++t) {
    const int d = t * 16 + g;
    const float vs = vsum[bh * HD_ + d];
#pragma unroll
    for (int j = 0; j < 8; ++j) {
      const int r = row0 + j + 8 * hi;
      const float o = (oacc[t][j] + epsn * vs) / (l[j] + EPS_);
      oh[((size_t)(b * N_ + r)) * C_ + h * HD_ + d] = (_Float16)o;
    }
  }
}

// ---------------------------------------------------------------- projection GEMM + bias
__global__ __launch_bounds__(256) void proj_gemm(const _Float16* __restrict__ A,
                                                 const _Float16* __restrict__ W,
                                                 const float* __restrict__ bias,
                                                 float* __restrict__ out) {
  const int tid = threadIdx.x, wave = tid >> 5, lane = tid & 31;
  const int g = lane & 15, hi = lane >> 4;
  const int mw = wave >> 1, nw = wave & 1;
  const int row0 = blockIdx.x * 128 + mw * 32;
  const int col0 = blockIdx.y * 128 + nw * 64;

  v8f acc[2][4] = {};
  for (int k0 = 0; k0 < C_; k0 += 32) {
    v16h a0 = load_afrag(A, C_, row0,      k0, g, hi);
    v16h a1 = load_afrag(A, C_, row0 + 16, k0, g, hi);
#pragma unroll
    for (int t = 0; t < 4; ++t) {
      v16h b = load_bfrag(W, C_, col0 + t * 16, k0, g, hi);
      acc[0][t] = WMMA_F16(a0, b, acc[0][t]);
      acc[1][t] = WMMA_F16(a1, b, acc[1][t]);
    }
  }
#pragma unroll
  for (int ms = 0; ms < 2; ++ms)
#pragma unroll
    for (int t = 0; t < 4; ++t) {
      const int col = col0 + t * 16 + g;
      const float bv = bias[col];
#pragma unroll
      for (int j = 0; j < 8; ++j) {
        const int row = row0 + ms * 16 + j + 8 * hi;
        out[(size_t)row * C_ + col] = acc[ms][t][j] + bv;
      }
    }
}

extern "C" void kernel_launch(void* const* d_in, const int* in_sizes, int n_in,
                              void* d_out, int out_size, void* d_ws, size_t ws_size,
                              hipStream_t stream) {
  const float* x      = (const float*)d_in[0];   // [8,1024,768]
  const float* policy = (const float*)d_in[1];   // [8,1024,1]
  const float* qkv_w  = (const float*)d_in[2];   // [2304,768]
  const float* proj_w = (const float*)d_in[3];   // [768,768]
  const float* proj_b = (const float*)d_in[4];   // [768]
  float* out = (float*)d_out;

  char* ws = (char*)d_ws;
  size_t off = 0;
  _Float16* xh    = (_Float16*)(ws + off); off += (size_t)B_ * N_ * C_ * 2;
  _Float16* wqkv  = (_Float16*)(ws + off); off += (size_t)3 * C_ * C_ * 2;
  _Float16* wproj = (_Float16*)(ws + off); off += (size_t)C_ * C_ * 2;
  _Float16* qh    = (_Float16*)(ws + off); off += (size_t)B_ * H_ * N_ * HD_ * 2;
  _Float16* kh    = (_Float16*)(ws + off); off += (size_t)B_ * H_ * N_ * HD_ * 2;
  _Float16* vT    = (_Float16*)(ws + off); off += (size_t)B_ * H_ * N_ * HD_ * 2;
  _Float16* oh    = (_Float16*)(ws + off); off += (size_t)B_ * N_ * C_ * 2;
  float*    vsum  = (float*)(ws + off);    off += (size_t)B_ * H_ * HD_ * 4;

  cvt_f32_f16<<<1024, 256, 0, stream>>>(x,      xh,    B_ * N_ * C_);
  cvt_f32_f16<<<512,  256, 0, stream>>>(qkv_w,  wqkv,  3 * C_ * C_);
  cvt_f32_f16<<<256,  256, 0, stream>>>(proj_w, wproj, C_ * C_);

  qkv_gemm<<<dim3(B_ * N_ / 128, 3 * C_ / 128), 256, 0, stream>>>(xh, wqkv, qh, kh, vT);

  vsum_kernel<<<B_ * H_, HD_, 0, stream>>>(vT, vsum);

  attn_kernel<<<(B_ * H_ * (N_ / 16)) / 8, 256, 0, stream>>>(qh, kh, vT, policy, vsum, oh);

  proj_gemm<<<dim3(B_ * N_ / 128, C_ / 128), 256, 0, stream>>>(oh, wproj, proj_b, out);
}
